// GAT_layer_50457275793871
// MI455X (gfx1250) — matrix operand order repaired and checked by reading, not compiled, not run on hardware
//
#include <hip/hip_runtime.h>
#include <hip/hip_bf16.h>

typedef __attribute__((ext_vector_type(16))) __bf16 v16bf;
typedef __attribute__((ext_vector_type(8)))  float  v8f;
typedef __attribute__((ext_vector_type(4)))  unsigned int v4u;
typedef __attribute__((ext_vector_type(8)))  int v8i;
typedef __attribute__((ext_vector_type(4)))  int v4i;

#define N_NODES 8192
#define D_FEAT  256
#define NEG_BIG (-9.0e15f)
#define ATT_SCALE 0.0625f   // 1/sqrt(256)

// LDS layout (dynamic): Hj[2][64*256] | Vt[2][256*64] | Pbuf[4][16*64] | Ab[4][16*256]
#define SM_HJ   0
#define SM_VT   65536
#define SM_PB   131072
#define SM_AB   139264
#define SM_SIZE (139264 + 32768)

union BF16Frag {
    v16bf v;
    uint4 q[2];
};

// ---------------------------------------------------------------- TDM 2D load
// Build a D# (ISA 8.3/8.4) for a 2D tile and issue TENSOR_LOAD_TO_LDS.
// Dims/strides in 2-byte elements; lds_addr/global addr in bytes.
__device__ static inline void tdm_load_2d(unsigned lds_addr, const void* gaddr,
                                          unsigned tensor_d0, unsigned tensor_d1,
                                          unsigned tile_d0, unsigned tile_d1,
                                          unsigned stride0) {
    const unsigned long long ga = (unsigned long long)gaddr;
    v4u g0;
    g0[0] = 1u;                                            // count=1 (valid), user mode
    g0[1] = lds_addr;                                      // LDS byte address
    g0[2] = (unsigned)ga;                                  // global_addr[31:0]
    g0[3] = (unsigned)((ga >> 32) & 0x01FFFFFFu) | (2u << 30);  // ga[56:32] | type=2
    v8i g1;
    g1[0] = (int)(1u << 16);                               // data_size=1 -> 2 bytes
    g1[1] = (int)((tensor_d0 & 0xFFFFu) << 16);            // tensor_dim0[15:0]
    g1[2] = (int)((tensor_d0 >> 16) | ((tensor_d1 & 0xFFFFu) << 16));
    g1[3] = (int)((tensor_d1 >> 16) | (tile_d0 << 16));    // tensor_dim1[31:16]|tile_dim0
    g1[4] = (int)(tile_d1 & 0xFFFFu);                      // tile_dim1, tile_dim2=0
    g1[5] = (int)stride0;                                  // tensor_dim0_stride[31:0]
    g1[6] = 0;                                             // stride0[47:32], stride1[15:0]
    g1[7] = 0;
    const v4i z4 = {0, 0, 0, 0};                           // groups 2/3: NULL (2D tensor)
    const v8i z8 = {0, 0, 0, 0, 0, 0, 0, 0};
    __builtin_amdgcn_tensor_load_to_lds(g0, g1, z4, z4, z8, 0);
}

// ---------------------------------------------------------------- fused GAT main
// Block: 128 threads = 4 waves; each wave owns a 16-row strip (64 rows/block).
// B-operand tiles (h_j, hnewT) staged by the Tensor Data Mover into LDS,
// double-buffered: TDM loads tile j+1 while waves compute tile j.
// A-strip parked in LDS to keep VGPR allocation below 256.
__global__ void __launch_bounds__(128)
k_gat(const __bf16* __restrict__ h_bf, const int* __restrict__ adj,
      const float* __restrict__ hnew, const __bf16* __restrict__ hnewT,
      const int* __restrict__ k_ptr, float* __restrict__ out) {
    extern __shared__ char smem[];
    __bf16* Hj = (__bf16*)(smem + SM_HJ);   // [2][64 rows][256 k]  bf16
    __bf16* Vt = (__bf16*)(smem + SM_VT);   // [2][256 cols][64 j]  bf16
    __bf16* Pb = (__bf16*)(smem + SM_PB);   // [4 waves][16 rows][64 j]
    __bf16* Ab = (__bf16*)(smem + SM_AB);   // [4 waves][16 rows][256 k]

    const int tid  = threadIdx.x;
    const int lane = tid & 31;
    const int wave = tid >> 5;
    const int half = lane >> 4;
    const int l16  = lane & 15;
    const int i0   = (blockIdx.x * 4 + wave) * 16;
    __bf16* Pw = Pb + wave * (16 * 64);
    __bf16* Aw = Ab + wave * (16 * 256);

    const unsigned ldsHj = (unsigned)(size_t)(void*)Hj;
    const unsigned ldsVt = (unsigned)(size_t)(void*)Vt;

    // stage tile at column jn into buffer buf via TDM (wave 0 only)
    auto stage = [&](int jn, int buf) {
        // Hj: rows [jn, jn+64) of h_bf -> linear 32KB (tile 256 x 64, stride 256)
        tdm_load_2d(ldsHj + buf * 32768u, h_bf + (size_t)jn * D_FEAT,
                    256u, 64u, 256u, 64u, 256u);
        // Vt: 256 rows x 64 cols of hnewT (stride 8192) packed into [256][64]
        tdm_load_2d(ldsVt + buf * 32768u, hnewT + jn,
                    64u, 256u, 64u, 256u, 8192u);
    };

    // ---- park this wave's A strip (16x256 bf16, 8KB) in its LDS region
    {
        const int r = lane >> 1, c = (lane & 1) * 128;
        const __bf16* src = h_bf + (size_t)(i0 + r) * D_FEAT + c;
        __bf16* dst = Aw + r * 256 + c;
#pragma unroll
        for (int q = 0; q < 16; ++q)
            *(uint4*)(dst + q * 8) = *(const uint4*)(src + q * 8);
    }

    v8f acc[16] = {};          // 16 rows x 256 cols f32 accumulator
    float m[8], l[8];
#pragma unroll
    for (int r = 0; r < 8; ++r) { m[r] = -3.0e38f; l[r] = 0.0f; }

    const size_t adj_row0 = (size_t)(i0 + half * 8) * N_NODES;

    if (wave == 0) {
        stage(0, 0);
        __builtin_amdgcn_s_wait_tensorcnt(0);
    }
    __syncthreads();
    int cur = 0;

    for (int j0 = 0; j0 < N_NODES; j0 += 64) {
        if (j0 + 64 < N_NODES) {
            if (wave == 0) stage(j0 + 64, cur ^ 1);        // TDM overlaps compute
            __builtin_prefetch(&adj[adj_row0 + j0 + 64 + l16], 0, 3);
        }

        // ---- reload A fragments from LDS (live only through the S phase)
        BF16Frag a[8];
        {
            const __bf16* ap = Aw + l16 * 256 + half * 8;
#pragma unroll
            for (int kk = 0; kk < 8; ++kk) {
                a[kk].q[0] = *(const uint4*)(ap + kk * 32);
                a[kk].q[1] = *(const uint4*)(ap + kk * 32 + 16);
            }
        }

        // ---- S = (h_i . h_j^T) * scale, masked by adj (B-frags from LDS,
        //      software-pipelined: load kk+1 while WMMA kk executes)
        float s[4][8];
#pragma unroll
        for (int tn = 0; tn < 4; ++tn) {
            v8f sv = {};
            const __bf16* bp0 = Hj + cur * 16384 + (tn * 16 + l16) * 256 + half * 8;
            BF16Frag b0, b1;
            b0.q[0] = *(const uint4*)(bp0);
            b0.q[1] = *(const uint4*)(bp0 + 16);
#pragma unroll
            for (int kk = 0; kk < 8; ++kk) {
                if (kk < 7) {
                    b1.q[0] = *(const uint4*)(bp0 + (kk + 1) * 32);
                    b1.q[1] = *(const uint4*)(bp0 + (kk + 1) * 32 + 16);
                }
                sv = __builtin_amdgcn_wmma_f32_16x16x32_bf16(false, a[kk].v, false, b0.v,
                                                             (short)0, sv, false, false);
                if (kk < 7) b0 = b1;
            }
#pragma unroll
            for (int r = 0; r < 8; ++r) {
                const int aj = adj[adj_row0 + (size_t)r * N_NODES + j0 + tn * 16 + l16];
                s[tn][r] = (aj > 0) ? sv[r] * ATT_SCALE : NEG_BIG;
            }
        }

        // ---- online softmax (row stats live in 16-lane halves)
        float fac[8];
#pragma unroll
        for (int r = 0; r < 8; ++r) {
            float v = fmaxf(fmaxf(s[0][r], s[1][r]), fmaxf(s[2][r], s[3][r]));
            v = fmaxf(v, __shfl_xor(v, 1, 32));
            v = fmaxf(v, __shfl_xor(v, 2, 32));
            v = fmaxf(v, __shfl_xor(v, 4, 32));
            v = fmaxf(v, __shfl_xor(v, 8, 32));
            const float mn = fmaxf(m[r], v);
            fac[r] = __expf(m[r] - mn);
            m[r] = mn;
            float ps = 0.0f;
#pragma unroll
            for (int tn = 0; tn < 4; ++tn) {
                const float p = __expf(s[tn][r] - mn);
                s[tn][r] = p;
                ps += p;
            }
            ps += __shfl_xor(ps, 1, 32);
            ps += __shfl_xor(ps, 2, 32);
            ps += __shfl_xor(ps, 4, 32);
            ps += __shfl_xor(ps, 8, 32);
            l[r] = l[r] * fac[r] + ps;
        }
#pragma unroll
        for (int ct = 0; ct < 16; ++ct)
#pragma unroll
            for (int r = 0; r < 8; ++r)
                acc[ct][r] *= fac[r];

        // ---- stage P (bf16) D-layout -> row-major LDS (same wave, no barrier)
#pragma unroll
        for (int tn = 0; tn < 4; ++tn)
#pragma unroll
            for (int r = 0; r < 8; ++r)
                Pw[(r + half * 8) * 64 + tn * 16 + l16] = (__bf16)s[tn][r];

        // ---- acc += P @ h_new[j0:j0+64, :]  (P and V frags from LDS,
        //      software-pipelined over ct)
#pragma unroll
        for (int kk2 = 0; kk2 < 2; ++kk2) {
            BF16Frag p;
            const __bf16* pp = Pw + l16 * 64 + kk2 * 32 + half * 8;
            p.q[0] = *(const uint4*)pp;
            p.q[1] = *(const uint4*)(pp + 16);
            const __bf16* vb = Vt + cur * 16384 + l16 * 64 + kk2 * 32 + half * 8;
            BF16Frag b0, b1;
            b0.q[0] = *(const uint4*)vb;
            b0.q[1] = *(const uint4*)(vb + 16);
#pragma unroll
            for (int ct = 0; ct < 16; ++ct) {
                if (ct < 15) {
                    const __bf16* vp = vb + (ct + 1) * 16 * 64;
                    b1.q[0] = *(const uint4*)vp;
                    b1.q[1] = *(const uint4*)(vp + 16);
                }
                acc[ct] = __builtin_amdgcn_wmma_f32_16x16x32_bf16(false, p.v, false, b0.v,
                                                                  (short)0, acc[ct], false, false);
                if (ct < 15) b0 = b1;
            }
        }

        if (wave == 0) __builtin_amdgcn_s_wait_tensorcnt(0);  // next tile landed
        __syncthreads();
        cur ^= 1;
    }

    // ---- epilogue: out = relu(0.5*softmax@hnew + 0.5*hnew), identity rows < k
    const int kcnt = *k_ptr;
    float linv[8];
#pragma unroll
    for (int r = 0; r < 8; ++r) linv[r] = 1.0f / l[r];
#pragma unroll
    for (int ct = 0; ct < 16; ++ct) {
        const int col = ct * 16 + l16;
#pragma unroll
        for (int r = 0; r < 8; ++r) {
            const int row = i0 + r + half * 8;
            const float hv = hnew[(size_t)row * D_FEAT + col];
            const float v = (row < kcnt) ? hv
                                         : (0.5f * acc[ct][r] * linv[r] + 0.5f * hv);
            out[(size_t)row * D_FEAT + col] = fmaxf(v, 0.0f);
        }
    }
}

// ---------------------------------------------------------------- convert
__global__ void k_convert(const float* __restrict__ h, const float* __restrict__ W,
                          __bf16* __restrict__ h_bf, __bf16* __restrict__ W_bf) {
    int idx = blockIdx.x * blockDim.x + threadIdx.x;
    if (idx < N_NODES * D_FEAT) h_bf[idx] = (__bf16)h[idx];
    if (idx < D_FEAT * D_FEAT)  W_bf[idx] = (__bf16)W[idx];
}

// ---------------------------------------------------------------- k = nnz(adj[:,0])
__global__ void k_count(const int* __restrict__ adj, int* __restrict__ k_out) {
    __shared__ int red[256];
    int t = threadIdx.x;
    int s = 0;
    for (int i = t; i < N_NODES; i += 256)
        s += (adj[(size_t)i * N_NODES] != 0) ? 1 : 0;
    red[t] = s;
    __syncthreads();
    for (int o = 128; o > 0; o >>= 1) {
        if (t < o) red[t] += red[t + o];
        __syncthreads();
    }
    if (t == 0) *k_out = red[0];
}

// ---------------------------------------------------------------- h_new = h @ W^T + b
__global__ void __launch_bounds__(128)
k_hnew(const __bf16* __restrict__ h_bf, const __bf16* __restrict__ W_bf,
       const float* __restrict__ bias,
       float* __restrict__ hnew, __bf16* __restrict__ hnewT) {
    const int lane = threadIdx.x & 31;
    const int wave = threadIdx.x >> 5;
    const int half = lane >> 4;
    const int l16  = lane & 15;
    const int i0   = blockIdx.x * 16;

    v8f acc[4] = {};
    const int arow = i0 + l16;
    for (int kk = 0; kk < 8; ++kk) {
        BF16Frag a;
        const __bf16* ap = h_bf + (size_t)arow * D_FEAT + kk * 32 + half * 8;
        a.q[0] = *(const uint4*)ap;
        a.q[1] = *(const uint4*)(ap + 16);
#pragma unroll
        for (int c = 0; c < 4; ++c) {
            const int ct = wave * 4 + c;
            BF16Frag b;   // B[k][n] = W[n][k] : row of W, contiguous in k
            const __bf16* bp = W_bf + (size_t)(ct * 16 + l16) * D_FEAT + kk * 32 + half * 8;
            b.q[0] = *(const uint4*)bp;
            b.q[1] = *(const uint4*)(bp + 16);
            acc[c] = __builtin_amdgcn_wmma_f32_16x16x32_bf16(false, a.v, false, b.v,
                                                             (short)0, acc[c], false, false);
        }
    }
#pragma unroll
    for (int c = 0; c < 4; ++c) {
        const int ct  = wave * 4 + c;
        const int col = ct * 16 + l16;
        const float bv = bias[col];
#pragma unroll
        for (int r = 0; r < 8; ++r) {
            const int orow = i0 + r + half * 8;
            const float v = acc[c][r] + bv;
            hnew[(size_t)orow * D_FEAT + col] = v;
            hnewT[(size_t)col * N_NODES + orow] = (__bf16)v;
        }
    }
}

// ---------------------------------------------------------------- launch
extern "C" void kernel_launch(void* const* d_in, const int* in_sizes, int n_in,
                              void* d_out, int out_size, void* d_ws, size_t ws_size,
                              hipStream_t stream) {
    const float* h   = (const float*)d_in[0];
    const int*   adj = (const int*)d_in[1];
    const float* W   = (const float*)d_in[2];
    const float* b   = (const float*)d_in[3];
    float* out = (float*)d_out;

    char* ws = (char*)d_ws;
    __bf16* h_bf  = (__bf16*)(ws);                                  // 4 MB
    __bf16* W_bf  = (__bf16*)(ws + (size_t)4  * 1024 * 1024);       // 128 KB
    __bf16* hnewT = (__bf16*)(ws + (size_t)5  * 1024 * 1024);       // 4 MB
    float*  hnew  = (float*) (ws + (size_t)9  * 1024 * 1024);       // 8 MB
    int*    k_val = (int*)   (ws + (size_t)17 * 1024 * 1024);       // 4 B

    k_convert<<<(N_NODES * D_FEAT + 255) / 256, 256, 0, stream>>>(h, W, h_bf, W_bf);
    k_count<<<1, 256, 0, stream>>>(adj, k_val);
    k_hnew<<<N_NODES / 16, 128, 0, stream>>>(h_bf, W_bf, b, hnew, hnewT);
    k_gat<<<N_NODES / 64, 128, SM_SIZE, stream>>>(h_bf, adj, hnew, hnewT, k_val, out);
}
